// AttnBlock_58557584113924
// MI455X (gfx1250) — matrix operand order repaired and checked by reading, not compiled, not run on hardware
//
#include <hip/hip_runtime.h>
#include <stdint.h>

#define HW   9216
#define CCH  128
#define GSZ  (4 * HW)        // elements per groupnorm group (C/32 * HW)
#define EPSF 1e-6f

#define KSPLIT 4
#define KEYS_PER_SPLIT (HW / KSPLIT)    // 2304
#define KSTEPS (KEYS_PER_SPLIT / 32)    // 72
#define KSTRIDE 136                     // u16 stride for staged K rows (272B, conflict-free)
#define VSTRIDE 40                      // u16 stride for staged V rows (80B, conflict-free)
#define PSTRIDE 40

typedef __attribute__((ext_vector_type(16))) __bf16 v16bf;
typedef __attribute__((ext_vector_type(8)))  float  v8f;

union BF16x16 { uint4 u4[2]; v16bf v; };

__device__ __forceinline__ uint16_t f2bf(float f) {
    uint32_t u = __float_as_uint(f);
    u += 0x7FFFu + ((u >> 16) & 1u);   // round-to-nearest-even
    return (uint16_t)(u >> 16);
}

// A-matrix fragment (16x32 bf16): lane half 0 -> K 0..15, half 1 -> K 16..31 (contiguous)
__device__ __forceinline__ v16bf load_a_frag(const uint16_t* rowk, int half) {
    BF16x16 f;
    const uint16_t* p = rowk + half * 16;
    f.u4[0] = *reinterpret_cast<const uint4*>(p);
    f.u4[1] = *reinterpret_cast<const uint4*>(p + 8);
    return f.v;
}

// B-matrix fragment (32x16 bf16): lane half 0 -> K 0..15, half 1 -> K 16..31 (contiguous)
__device__ __forceinline__ v16bf load_b_frag(const uint16_t* rowk, int half) {
    BF16x16 f;
    const uint16_t* p = rowk + half * 16;
    f.u4[0] = *reinterpret_cast<const uint4*>(p);
    f.u4[1] = *reinterpret_cast<const uint4*>(p + 8);
    return f.v;
}

// gfx1250 async global->LDS copy: 16B per lane, tracked by ASYNCcnt
__device__ __forceinline__ void async_load_b128(uint32_t lds_off, const void* gptr) {
    asm volatile("global_load_async_to_lds_b128 %0, %1, off"
                 :: "v"(lds_off), "v"(gptr) : "memory");
}
__device__ __forceinline__ void wait_async0() {
    asm volatile("s_wait_asynccnt 0" ::: "memory");
}
__device__ __forceinline__ uint32_t lds_of(const void* p) {
    return (uint32_t)(uintptr_t)p;     // low 32 bits of a shared pointer = LDS byte address
}

// ---------------- GroupNorm statistics: 32 groups of 36864 elements ----------------
__global__ void gn_stats_kernel(const float* __restrict__ x, float* __restrict__ stats) {
    int g = blockIdx.x;
    const float* xg = x + (size_t)g * GSZ;
    float s = 0.f, s2 = 0.f;
    for (int i = threadIdx.x; i < GSZ; i += blockDim.x) {
        float v = xg[i];
        s += v; s2 += v * v;
    }
    __shared__ float rs[256], rs2[256];
    rs[threadIdx.x] = s; rs2[threadIdx.x] = s2;
    __syncthreads();
    for (int st = 128; st > 0; st >>= 1) {
        if ((int)threadIdx.x < st) {
            rs[threadIdx.x]  += rs[threadIdx.x + st];
            rs2[threadIdx.x] += rs2[threadIdx.x + st];
        }
        __syncthreads();
    }
    if (threadIdx.x == 0) {
        float mean = rs[0] / (float)GSZ;
        float var  = rs2[0] / (float)GSZ - mean * mean;
        stats[g]      = mean;
        stats[32 + g] = rsqrtf(var + EPSF);
    }
}

// ---------------- Convert the four 128x128 fp32 weights to bf16 ----------------
__global__ void wcvt_kernel(const float* __restrict__ wq, const float* __restrict__ wk,
                            const float* __restrict__ wv, const float* __restrict__ wo,
                            uint16_t* __restrict__ wqB, uint16_t* __restrict__ wkB,
                            uint16_t* __restrict__ wvB, uint16_t* __restrict__ woB) {
    int i = blockIdx.x * blockDim.x + threadIdx.x;
    if (i < CCH * CCH) {
        wqB[i] = f2bf(wq[i]); wkB[i] = f2bf(wk[i]);
        wvB[i] = f2bf(wv[i]); woB[i] = f2bf(wo[i]);
    }
}

// ---------------- Apply GroupNorm, store transposed bf16 hT[p][c] ----------------
__global__ void gn_apply_kernel(const float* __restrict__ x, const float* __restrict__ gamma,
                                const float* __restrict__ beta, const float* __restrict__ stats,
                                uint16_t* __restrict__ hT) {
    int idx = blockIdx.x * blockDim.x + threadIdx.x;
    if (idx >= CCH * HW) return;
    int c = idx / HW, p = idx - c * HW;
    int g = c >> 2;
    float v = (x[idx] - stats[g]) * stats[32 + g] * gamma[c] + beta[c];
    hT[p * CCH + c] = f2bf(v);
}

// ---------------- Fused Q/K/V projection (one 16x16 tile per wave, K=128) ----------------
__global__ void __launch_bounds__(128) qkv_kernel(
        const uint16_t* __restrict__ hT,
        const uint16_t* __restrict__ wqB, const uint16_t* __restrict__ wkB,
        const uint16_t* __restrict__ wvB,
        const float* __restrict__ bq, const float* __restrict__ bk, const float* __restrict__ bv,
        uint16_t* __restrict__ qb, uint16_t* __restrict__ kb, uint16_t* __restrict__ vT) {
    int wave = (blockIdx.x * blockDim.x + threadIdx.x) >> 5;
    int lane = threadIdx.x & 31;
    int half = lane >> 4, l16 = lane & 15;
    int o0 = (wave & 7) * 16;          // 8 output-channel tiles
    int p0 = (wave >> 3) * 16;         // 576 spatial tiles

    v8f aq = {}; v8f ak = {}; v8f av = {};
    const uint16_t* brow  = hT  + (size_t)(p0 + l16) * CCH;
    const uint16_t* aqrow = wqB + (size_t)(o0 + l16) * CCH;
    const uint16_t* akrow = wkB + (size_t)(o0 + l16) * CCH;
    const uint16_t* avrow = wvB + (size_t)(o0 + l16) * CCH;

#pragma unroll
    for (int kc = 0; kc < 4; ++kc) {
        v16bf bf = load_b_frag(brow + kc * 32, half);
        v16bf a0 = load_a_frag(aqrow + kc * 32, half);
        v16bf a1 = load_a_frag(akrow + kc * 32, half);
        v16bf a2 = load_a_frag(avrow + kc * 32, half);
        aq = __builtin_amdgcn_wmma_f32_16x16x32_bf16(false, a0, false, bf, (short)0, aq, false, false);
        ak = __builtin_amdgcn_wmma_f32_16x16x32_bf16(false, a1, false, bf, (short)0, ak, false, false);
        av = __builtin_amdgcn_wmma_f32_16x16x32_bf16(false, a2, false, bf, (short)0, av, false, false);
    }
#pragma unroll
    for (int r = 0; r < 8; ++r) {
        int o = o0 + r + half * 8;
        int p = p0 + l16;
        qb[(size_t)o * HW + p] = f2bf(aq[r] + bq[o]);
        kb[(size_t)o * HW + p] = f2bf(ak[r] + bk[o]);
        // raw-reshape aliasing: flat o*HW+p == j*128+c  ->  j = o*72 + p/128, c = p%128
        int j = o * 72 + (p >> 7);
        int c = p & 127;
        vT[(size_t)c * HW + j] = f2bf(av[r] + bv[o]);   // V transposed for the PV matmul
    }
}

// ---------------- Flash attention with 4-way key split + async LDS K/V staging ----------
// Block = 4 waves (128 thr). Each wave owns one 16-row query tile; the block shares one
// double-buffered K/V stream staged via global_load_async_to_lds_b128 (ASYNCcnt).
__device__ __forceinline__ void stage_kv(const uint16_t* __restrict__ kb,
                                         const uint16_t* __restrict__ vT,
                                         int j0, uint32_t kbase, uint32_t vbase, int tid) {
    // K block: 32 rows x 256B -> 512 x 16B chunks, 128 threads -> 4 rounds
#pragma unroll
    for (int r = 0; r < 4; ++r) {
        int chunk = r * 128 + tid;
        int row = chunk >> 4;
        int c8  = chunk & 15;
        async_load_b128(kbase + (uint32_t)(row * (KSTRIDE * 2) + c8 * 16),
                        kb + (size_t)(j0 + row) * CCH + c8 * 8);
    }
    // V block (vT slice): 128 rows x 64B -> 512 x 16B chunks
#pragma unroll
    for (int r = 0; r < 4; ++r) {
        int chunk = r * 128 + tid;
        int row = chunk >> 2;
        int c8  = chunk & 3;
        async_load_b128(vbase + (uint32_t)(row * (VSTRIDE * 2) + c8 * 16),
                        vT + (size_t)row * HW + j0 + c8 * 8);
    }
}

__global__ void __launch_bounds__(128) flash_split_kernel(
        const uint16_t* __restrict__ qb, const uint16_t* __restrict__ kb,
        const uint16_t* __restrict__ vT,
        float* __restrict__ Opart, float* __restrict__ Mpart, float* __restrict__ Lpart) {
    __shared__ __attribute__((aligned(16))) uint16_t kst[2][32 * KSTRIDE];
    __shared__ __attribute__((aligned(16))) uint16_t vst[2][128 * VSTRIDE];
    __shared__ __attribute__((aligned(16))) uint16_t pst[4][16 * PSTRIDE];

    int tid  = threadIdx.x;
    int wid  = tid >> 5;
    int lane = tid & 31;
    int half = lane >> 4, l16 = lane & 15;
    int qgroup = blockIdx.x >> 2;                  // 144 query groups of 4 tiles
    int split  = blockIdx.x & 3;                   // key-range split
    int i0     = (qgroup * 4 + wid) * 16;
    int jbase  = split * KEYS_PER_SPLIT;
    uint16_t* lds = pst[wid];

    // Q A-fragments, resident for the whole sweep (Qm is the (9216,128) raw view of qb)
    v16bf qf[4];
    const uint16_t* qrow = qb + (size_t)(i0 + l16) * CCH;
#pragma unroll
    for (int kc = 0; kc < 4; ++kc) qf[kc] = load_a_frag(qrow + kc * 32, half);

    v8f o_acc[8];
#pragma unroll
    for (int t = 0; t < 8; ++t) o_acc[t] = (v8f){};
    float m_r[8], l_r[8];
#pragma unroll
    for (int r = 0; r < 8; ++r) { m_r[r] = -1e30f; l_r[r] = 0.f; }
    const float scale = 0.08838834764831845f;      // 128^-0.5

    // prologue: stage first K/V block
    stage_kv(kb, vT, jbase, lds_of(&kst[0][0]), lds_of(&vst[0][0]), tid);
    wait_async0();
    __syncthreads();

    for (int it = 0; it < KSTEPS; ++it) {
        int cur = it & 1;
        if (it + 1 < KSTEPS)
            stage_kv(kb, vT, jbase + (it + 1) * 32,
                     lds_of(&kst[1 - cur][0]), lds_of(&vst[1 - cur][0]), tid);

        // S tiles for this 32-key block (K fragments from LDS)
        v8f s[2];
#pragma unroll
        for (int jj = 0; jj < 2; ++jj) {
            v8f acc = {};
            const uint16_t* krow = &kst[cur][(jj * 16 + l16) * KSTRIDE];
#pragma unroll
            for (int kc = 0; kc < 4; ++kc) {
                v16bf bf = load_b_frag(krow + kc * 32, half);
                acc = __builtin_amdgcn_wmma_f32_16x16x32_bf16(false, qf[kc], false, bf, (short)0, acc, false, false);
            }
            s[jj] = acc;
        }
        __syncthreads();
        // online softmax; row r lives in lanes of one 16-lane half (C/D layout)
#pragma unroll
        for (int r = 0; r < 8; ++r) {
            float s0 = s[0][r] * scale, s1 = s[1][r] * scale;
            float mx = fmaxf(s0, s1);
#pragma unroll
            for (int msk = 1; msk < 16; msk <<= 1) mx = fmaxf(mx, __shfl_xor(mx, msk, 32));
            float m_new = fmaxf(m_r[r], mx);
            float alpha = __expf(m_r[r] - m_new);
            float p0 = __expf(s0 - m_new);
            float p1 = __expf(s1 - m_new);
            float rs = p0 + p1;
#pragma unroll
            for (int msk = 1; msk < 16; msk <<= 1) rs += __shfl_xor(rs, msk, 32);
            l_r[r] = l_r[r] * alpha + rs;
            m_r[r] = m_new;
#pragma unroll
            for (int t = 0; t < 8; ++t) o_acc[t][r] *= alpha;
            int m = r + half * 8;
            lds[m * PSTRIDE + l16]      = f2bf(p0);   // P staged in A-layout [m][k]
            lds[m * PSTRIDE + 16 + l16] = f2bf(p1);
        }
        __syncthreads();
        // P A-fragment (16x32 over the 32 keys)
        BF16x16 pf;
        const uint16_t* lp = lds + l16 * PSTRIDE + half * 16;
        pf.u4[0] = *reinterpret_cast<const uint4*>(lp);
        pf.u4[1] = *reinterpret_cast<const uint4*>(lp + 8);
        // O += P @ V  (V fragments from LDS; 8 column chunks of 16 channels)
#pragma unroll
        for (int t = 0; t < 8; ++t) {
            const uint16_t* vrow = &vst[cur][(t * 16 + l16) * VSTRIDE + half * 16];
            BF16x16 bf;
            bf.u4[0] = *reinterpret_cast<const uint4*>(vrow);
            bf.u4[1] = *reinterpret_cast<const uint4*>(vrow + 8);
            o_acc[t] = __builtin_amdgcn_wmma_f32_16x16x32_bf16(false, pf.v, false, bf.v, (short)0, o_acc[t], false, false);
        }
        wait_async0();         // our async loads into the other buffer are done
        __syncthreads();       // everyone's loads landed; everyone done reading cur
    }

    // epilogue: write un-normalized partial O and per-row (m, l)
    size_t obase = ((size_t)split * HW + i0) * CCH;
#pragma unroll
    for (int t = 0; t < 8; ++t) {
#pragma unroll
        for (int r = 0; r < 8; ++r) {
            int irow = r + half * 8;
            int c = t * 16 + l16;
            Opart[obase + (size_t)irow * CCH + c] = o_acc[t][r];
        }
    }
    if (l16 == 0) {
#pragma unroll
        for (int r = 0; r < 8; ++r) {
            int i = i0 + r + half * 8;
            Mpart[split * HW + i] = m_r[r];
            Lpart[split * HW + i] = l_r[r];
        }
    }
}

// ---------------- Merge the 4 key-split partials (log-sum-exp), store transposed oT ---
__global__ void merge_kernel(const float* __restrict__ Opart, const float* __restrict__ Mpart,
                             const float* __restrict__ Lpart, uint16_t* __restrict__ oT) {
    int idx = blockIdx.x * blockDim.x + threadIdx.x;
    if (idx >= CCH * HW) return;
    int i = idx >> 7, c = idx & 127;
    float m[KSPLIT];
    float M = -1e30f;
#pragma unroll
    for (int s = 0; s < KSPLIT; ++s) { m[s] = Mpart[s * HW + i]; M = fmaxf(M, m[s]); }
    float L = 0.f, O = 0.f;
#pragma unroll
    for (int s = 0; s < KSPLIT; ++s) {
        float w = __expf(m[s] - M);
        L += w * Lpart[s * HW + i];
        O += w * Opart[((size_t)s * HW + i) * CCH + c];
    }
    float val = O / L;
    int cch = i / 72;                   // inverse raw-reshape: flat i*128+c == cch*9216 + pp
    int pp  = (i - cch * 72) * 128 + c;
    oT[(size_t)pp * CCH + cch] = f2bf(val);
}

// ---------------- Output projection + bias + residual ----------------
__global__ void __launch_bounds__(128) oproj_kernel(
        const uint16_t* __restrict__ oT, const uint16_t* __restrict__ woB,
        const float* __restrict__ bo, const float* __restrict__ x, float* __restrict__ out) {
    int wave = (blockIdx.x * blockDim.x + threadIdx.x) >> 5;
    int lane = threadIdx.x & 31;
    int half = lane >> 4, l16 = lane & 15;
    int o0 = (wave & 7) * 16;
    int p0 = (wave >> 3) * 16;

    v8f acc = {};
    const uint16_t* brow = oT  + (size_t)(p0 + l16) * CCH;
    const uint16_t* arow = woB + (size_t)(o0 + l16) * CCH;
#pragma unroll
    for (int kc = 0; kc < 4; ++kc) {
        v16bf bf = load_b_frag(brow + kc * 32, half);
        v16bf af = load_a_frag(arow + kc * 32, half);
        acc = __builtin_amdgcn_wmma_f32_16x16x32_bf16(false, af, false, bf, (short)0, acc, false, false);
    }
#pragma unroll
    for (int r = 0; r < 8; ++r) {
        int o = o0 + r + half * 8;
        int p = p0 + l16;
        size_t idx = (size_t)o * HW + p;
        out[idx] = acc[r] + bo[o] + x[idx];
    }
}

extern "C" void kernel_launch(void* const* d_in, const int* in_sizes, int n_in,
                              void* d_out, int out_size, void* d_ws, size_t ws_size,
                              hipStream_t stream) {
    const float* x     = (const float*)d_in[0];
    const float* gamma = (const float*)d_in[1];
    const float* beta  = (const float*)d_in[2];
    const float* wq    = (const float*)d_in[3];
    const float* bq    = (const float*)d_in[4];
    const float* wk    = (const float*)d_in[5];
    const float* bk    = (const float*)d_in[6];
    const float* wv    = (const float*)d_in[7];
    const float* bv    = (const float*)d_in[8];
    const float* wo    = (const float*)d_in[9];
    const float* bo    = (const float*)d_in[10];
    float* out = (float*)d_out;

    const size_t NCHW = (size_t)CCH * HW;          // 1,179,648
    char* ws = (char*)d_ws;
    float*    stats = (float*)ws;                   // 64 floats
    uint16_t* wqB = (uint16_t*)(ws + 512);
    uint16_t* wkB = wqB + CCH * CCH;
    uint16_t* wvB = wkB + CCH * CCH;
    uint16_t* woB = wvB + CCH * CCH;
    uint16_t* hT  = woB + CCH * CCH;                // normalized, transposed bf16
    uint16_t* qb  = hT + NCHW;
    uint16_t* kb  = qb + NCHW;
    uint16_t* vT  = kb + NCHW;
    uint16_t* oT  = vT + NCHW;
    float* Opart  = (float*)(oT + NCHW);            // 4 x (9216 x 128) fp32 partials
    float* Mpart  = Opart + (size_t)KSPLIT * NCHW;
    float* Lpart  = Mpart + (size_t)KSPLIT * HW;

    gn_stats_kernel<<<32, 256, 0, stream>>>(x, stats);
    wcvt_kernel<<<(CCH * CCH + 255) / 256, 256, 0, stream>>>(wq, wk, wv, wo, wqB, wkB, wvB, woB);
    gn_apply_kernel<<<(int)((NCHW + 255) / 256), 256, 0, stream>>>(x, gamma, beta, stats, hT);
    qkv_kernel<<<1152, 128, 0, stream>>>(hT, wqB, wkB, wvB, bq, bk, bv, qb, kb, vT);
    flash_split_kernel<<<576, 128, 0, stream>>>(qb, kb, vT, Opart, Mpart, Lpart);
    merge_kernel<<<(int)((NCHW + 255) / 256), 256, 0, stream>>>(Opart, Mpart, Lpart, oT);
    oproj_kernel<<<1152, 128, 0, stream>>>(oT, woB, bo, x, out);
}
